// SwinTransformerBlock_23605140259511
// MI455X (gfx1250) — matrix operand order repaired and checked by reading, not compiled, not run on hardware
//
#include <hip/hip_runtime.h>

// ---------------------------------------------------------------------------
// Swin block for MI455X (gfx1250, wave32).  All GEMMs + attention use
// V_WMMA_F32_16X16X32_F16 (f16 A/B, f32 accumulate).  Activations staged as
// f16 in workspace to halve HBM traffic (workload is near the bandwidth roof).
// GEMM A-tiles are staged to LDS with GLOBAL_LOAD_ASYNC_TO_LDS_B128 (ASYNCcnt).
// ---------------------------------------------------------------------------

typedef __attribute__((ext_vector_type(16))) _Float16 v16h;
typedef __attribute__((ext_vector_type(8)))  float    v8f;

#define TOK     25088   // total window tokens = 512 windows * 49  (== B*H*W)
#define CDIM    384
#define HID_DIM 1536
#define QKV_DIM 1152
#define NWIN    512
#define HEADS   12
#define HDIM    32
#define NTOK    49
#define PTOK    64      // padded tokens per window-head for q/k/v

// async memory->LDS copy of 16 bytes, per-lane addresses (GV mode).
// Tracked by ASYNCcnt; fence with s_wait_asynccnt before the barrier.
#define ASYNC_COPY_B128(ldsPtr, gPtr)                                         \
  asm volatile("global_load_async_to_lds_b128 %0, %1, off"                    \
               :: "v"((unsigned)(size_t)(ldsPtr)),                            \
                  "v"((unsigned long long)(size_t)(gPtr))                     \
               : "memory")
#define ASYNC_WAIT() asm volatile("s_wait_asynccnt 0x0" ::: "memory")

__device__ inline v8f wmma_f16(v16h a, v16h b, v8f c) {
  return __builtin_amdgcn_wmma_f32_16x16x32_f16(false, a, false, b, (short)0, c,
                                                false, false);
}

// A fragment 16x32 (f16) from LDS, row-major with given half-stride.
// Layout: lanes0-15 row M=lane, K=0..15 (pairs per VGPR);
// lanes16-31 same rows, K=16..31.
__device__ inline v16h frag_a_lds(const _Float16* base, int stride, int lane) {
  int half_sel = lane >> 4, row = lane & 15;
  const _Float16* p = base + row * stride;
  v16h out;
#pragma unroll
  for (int r = 0; r < 8; ++r) {
    int kb = half_sel * 16 + r * 2;
    union { unsigned u; _Float16 f[2]; } cv;
    cv.u = *(const unsigned*)(p + kb);
    out[2 * r] = cv.f[0];
    out[2 * r + 1] = cv.f[1];
  }
  return out;
}

// B fragment 32x16 (f16) from LDS stored TRANSPOSED as [n][k] with stride.
// Layout used here (contiguous k, as in frag_a_lds): lanes0-15 N=lane, K=0..15 (pairs per VGPR); lanes16-31 K=16..31.
__device__ inline v16h frag_b_ldsT(const _Float16* base, int stride, int lane) {
  int n = lane & 15, koff = (lane >> 4) * 16;
  const _Float16* p = base + n * stride + koff;
  v16h out;
#pragma unroll
  for (int r = 0; r < 8; ++r) {
    union { unsigned u; _Float16 f[2]; } cv;
    cv.u = *(const unsigned*)(p + 2 * r);
    out[2 * r] = cv.f[0];
    out[2 * r + 1] = cv.f[1];
  }
  return out;
}

// ---------------------------------------------------------------------------
// Zero the padding rows (49..63) of the padded q/k/v buffers.
// Contiguous region per window-head: 480 halves = 240 dwords at offset
// wh*2048 + 1568 halves.
// ---------------------------------------------------------------------------
__global__ __launch_bounds__(256)
void zero_pad_kernel(unsigned* __restrict__ qb, unsigned* __restrict__ kb,
                     unsigned* __restrict__ vb) {
  int idx = blockIdx.x * 256 + threadIdx.x;          // 6144 * 240 dwords
  if (idx < NWIN * HEADS * 240) {
    int wh = idx / 240, o = idx % 240;
    size_t p = (size_t)wh * (PTOK * HDIM / 2) + (NTOK * HDIM / 2) + o;
    qb[p] = 0u; kb[p] = 0u; vb[p] = 0u;
  }
}

// ---------------------------------------------------------------------------
// LayerNorm over C=384, output f16.  mode 0: gather with cyclic shift (-3,-3)
// + window partition (row m = window-token order).  mode 1: identity rows.
// ---------------------------------------------------------------------------
__global__ __launch_bounds__(128)
void ln_kernel(const float* __restrict__ src, const float* __restrict__ g,
               const float* __restrict__ bta, _Float16* __restrict__ dst,
               int mode) {
  __shared__ float red[128];
  int m = blockIdx.x, tid = threadIdx.x;
  int srow;
  if (mode == 0) {
    int win = m / NTOK, tok = m % NTOK;
    int bi = win >> 6, wsp = win & 63;
    int wh = wsp >> 3, ww = wsp & 7;
    int i = tok / 7, j = tok % 7;
    int hh = (wh * 7 + i + 3) % 56;   // roll(-3): shifted[hs] = orig[(hs+3)%56]
    int wc = (ww * 7 + j + 3) % 56;
    srow = bi * 3136 + hh * 56 + wc;
  } else {
    srow = m;
  }
  const float* row = src + (size_t)srow * CDIM;
  float s = 0.f;
  for (int c = tid; c < CDIM; c += 128) s += row[c];
  red[tid] = s; __syncthreads();
  for (int o = 64; o > 0; o >>= 1) { if (tid < o) red[tid] += red[tid + o]; __syncthreads(); }
  float mu = red[0] * (1.0f / CDIM);
  __syncthreads();
  float v = 0.f;
  for (int c = tid; c < CDIM; c += 128) { float d = row[c] - mu; v += d * d; }
  red[tid] = v; __syncthreads();
  for (int o = 64; o > 0; o >>= 1) { if (tid < o) red[tid] += red[tid + o]; __syncthreads(); }
  float rinv = rsqrtf(red[0] * (1.0f / CDIM) + 1e-5f);
  for (int c = tid; c < CDIM; c += 128)
    dst[(size_t)m * CDIM + c] = (_Float16)((row[c] - mu) * rinv * g[c] + bta[c]);
}

// ---------------------------------------------------------------------------
// Generic WMMA GEMM: C[M,N] = A(f16)[M,K] * B(f32->f16)[K,N] + bias, with a
// mode-specific fused epilogue.  Block = 256 thr (8 waves), tile 128x128,
// each wave computes 2x4 16x16 tiles, K stepped by 32.
//   MODE 0: QKV  -> scatter to q(scaled)/k/v f16 [win*12+h][64(tok)][32]
//   MODE 1: proj -> window-reverse + roll(+3,+3) + residual(x) -> xr (f32)
//   MODE 2: FC1  -> exact GELU -> f16
//   MODE 3: FC2  -> + residual(xr) -> f32 d_out
// ---------------------------------------------------------------------------
template <int MODE>
__global__ __launch_bounds__(256)
void gemm_kernel(const _Float16* __restrict__ A, const float* __restrict__ Bw,
                 const float* __restrict__ bias, int M, int N, int K,
                 _Float16* __restrict__ outh, float* __restrict__ outf,
                 const float* __restrict__ res, _Float16* __restrict__ qb,
                 _Float16* __restrict__ kb, _Float16* __restrict__ vb) {
  __shared__ _Float16 lA[128 * 40];   // [row][k]  stride 40 halves (80B, 16B-mult)
  __shared__ _Float16 lB[128 * 40];   // [n][k] transposed, stride 40
  int tid = threadIdx.x, lane = tid & 31, wave = tid >> 5;
  int wm = wave & 3, wn = wave >> 2;            // wave grid 4(M) x 2(N)
  int m0 = blockIdx.y * 128, n0 = blockIdx.x * 128;
  v8f zero = {0.f, 0.f, 0.f, 0.f, 0.f, 0.f, 0.f, 0.f};
  v8f acc[2][4];
#pragma unroll
  for (int a = 0; a < 2; ++a)
#pragma unroll
    for (int b2 = 0; b2 < 4; ++b2) acc[a][b2] = zero;

  for (int k0 = 0; k0 < K; k0 += 32) {
    // stage A tile 128x32 f16 with async memory->LDS (16B per lane per issue)
#pragma unroll
    for (int i = 0; i < 2; ++i) {
      int lin = tid + i * 256;
      int r = lin >> 2, cg = lin & 3;
      ASYNC_COPY_B128(lA + r * 40 + cg * 8,
                      A + (size_t)(m0 + r) * K + k0 + cg * 8);
    }
    // stage B tile 32x128: f32 global -> f16 transposed LDS (needs conversion)
#pragma unroll
    for (int i = 0; i < 4; ++i) {
      int lin = tid + i * 256;
      int kk = lin >> 5, c4 = lin & 31;
      float4 f = *(const float4*)(Bw + (size_t)(k0 + kk) * N + n0 + c4 * 4);
      int nb = c4 * 4;
      lB[(nb + 0) * 40 + kk] = (_Float16)f.x;
      lB[(nb + 1) * 40 + kk] = (_Float16)f.y;
      lB[(nb + 2) * 40 + kk] = (_Float16)f.z;
      lB[(nb + 3) * 40 + kk] = (_Float16)f.w;
    }
    if (k0 + 32 < K)  // hint next A tile into cache (global_prefetch_b8)
      __builtin_prefetch(A + (size_t)(m0 + (tid >> 1)) * K + k0 + 32, 0, 0);
    ASYNC_WAIT();     // this wave's async LDS writes are complete
    __syncthreads();  // ... and visible to all waves

    v16h af[2], bf[4];
#pragma unroll
    for (int tm = 0; tm < 2; ++tm)
      af[tm] = frag_a_lds(lA + (wm * 32 + tm * 16) * 40, 40, lane);
#pragma unroll
    for (int tn = 0; tn < 4; ++tn)
      bf[tn] = frag_b_ldsT(lB + (wn * 64 + tn * 16) * 40, 40, lane);
#pragma unroll
    for (int tm = 0; tm < 2; ++tm)
#pragma unroll
      for (int tn = 0; tn < 4; ++tn)
        acc[tm][tn] = wmma_f16(af[tm], bf[tn], acc[tm][tn]);
    __syncthreads();
  }

  // epilogue: C/D layout -> m = half_sel*8 + vgpr, n = lane&15
  int half_sel = lane >> 4, ncol = lane & 15;
#pragma unroll
  for (int tm = 0; tm < 2; ++tm)
    for (int tn = 0; tn < 4; ++tn)
#pragma unroll
      for (int r = 0; r < 8; ++r) {
        int m = m0 + wm * 32 + tm * 16 + half_sel * 8 + r;
        int n = n0 + wn * 64 + tn * 16 + ncol;
        float val = acc[tm][tn][r] + bias[n];
        if (MODE == 0) {
          int s = n / CDIM, rem = n % CDIM, hh = rem >> 5, d = rem & 31;
          if (s == 0) val *= 0.17677669529663687f;   // HD^-0.5
          int win = m / NTOK, tok = m % NTOK;
          _Float16* dst = (s == 0) ? qb : ((s == 1) ? kb : vb);
          dst[(((size_t)win * HEADS + hh) * PTOK + tok) * HDIM + d] = (_Float16)val;
        } else if (MODE == 1) {
          int win = m / NTOK, tok = m % NTOK;
          int bi = win >> 6, wsp = win & 63, wh = wsp >> 3, ww = wsp & 7;
          int i = tok / 7, j = tok % 7;
          int hh2 = (wh * 7 + i + 3) % 56;   // roll(+3): final[(hs+3)%56]=shifted[hs]
          int wc2 = (ww * 7 + j + 3) % 56;
          size_t idx = ((size_t)bi * 3136 + hh2 * 56 + wc2) * CDIM + n;
          outf[idx] = res[idx] + val;
        } else if (MODE == 2) {
          val = 0.5f * val * (1.f + erff(val * 0.70710678118f));
          outh[(size_t)m * N + n] = (_Float16)val;
        } else {
          size_t idx = (size_t)m * N + n;
          outf[idx] = res[idx] + val;
        }
      }
}

// ---------------------------------------------------------------------------
// Windowed attention: one wave per (window, head).  q/k/v are zero-padded to
// 64 tokens so all fragment loads are unconditional (no exec-mask branching).
// S = q k^T (16 WMMAs) -> +rel-pos-bias +shift-mask -> row softmax (LDS) ->
// P(f16, LDS) -> O = P v (16 WMMAs) -> f16 out [win*49+tok][head*32+d].
// ---------------------------------------------------------------------------
__global__ __launch_bounds__(64)
void attn_kernel(const _Float16* __restrict__ qb, const _Float16* __restrict__ kb,
                 const _Float16* __restrict__ vb, const float* __restrict__ rpb,
                 const float* __restrict__ amask, _Float16* __restrict__ out) {
  __shared__ float    Sl[2][64 * 68];  // per-wave S scores (f32)
  __shared__ _Float16 Pl[2][64 * 72];  // per-wave probabilities (f16, zero-padded)
  int tid = threadIdx.x, lane = tid & 31, w = tid >> 5;
  int wh_idx = blockIdx.x * 2 + w;
  int win = wh_idx / HEADS, h = wh_idx % HEADS;
  const _Float16* q = qb + (size_t)wh_idx * PTOK * HDIM;
  const _Float16* k = kb + (size_t)wh_idx * PTOK * HDIM;
  const _Float16* v = vb + (size_t)wh_idx * PTOK * HDIM;
  int half_sel = lane >> 4, rowl = lane & 15, koff = half_sel * 16;
  v8f zero = {0.f, 0.f, 0.f, 0.f, 0.f, 0.f, 0.f, 0.f};

  // A = q tiles (M=token, K=d); padded rows are zero in memory
  v16h aq[4];
#pragma unroll
  for (int mi = 0; mi < 4; ++mi) {
    int tok = mi * 16 + rowl;
#pragma unroll
    for (int r = 0; r < 8; ++r) {
      int kbx = half_sel * 16 + r * 2;
      union { unsigned u; _Float16 f[2]; } cv;
      cv.u = *(const unsigned*)(q + tok * HDIM + kbx);
      aq[mi][2 * r] = cv.f[0];
      aq[mi][2 * r + 1] = cv.f[1];
    }
  }
  // B = k^T tiles (K=d, N=token)
  v16h bk[4];
#pragma unroll
  for (int nj = 0; nj < 4; ++nj) {
    int tokn = nj * 16 + rowl;
#pragma unroll
    for (int r = 0; r < 8; ++r) {
      union { unsigned u; _Float16 f[2]; } cv;
      cv.u = *(const unsigned*)(k + tokn * HDIM + koff + 2 * r);
      bk[nj][2 * r] = cv.f[0];
      bk[nj][2 * r + 1] = cv.f[1];
    }
  }
  float* S = &Sl[w][0];
  _Float16* P = &Pl[w][0];
#pragma unroll
  for (int mi = 0; mi < 4; ++mi)
#pragma unroll
    for (int nj = 0; nj < 4; ++nj) {
      v8f s = wmma_f16(aq[mi], bk[nj], zero);
#pragma unroll
      for (int r = 0; r < 8; ++r)
        S[(mi * 16 + half_sel * 8 + r) * 68 + nj * 16 + rowl] = s[r];
    }
  __syncthreads();

  for (int idx2 = lane; idx2 < 64 * 72; idx2 += 32) P[idx2] = (_Float16)0.f;
  const float* mrow = amask + (size_t)(win & 63) * NTOK * NTOK;
  for (int n = lane; n < NTOK; n += 32) {
    int rn = n / 7, cn = n % 7;
    float mx = -3.0e38f;
    for (int m = 0; m < NTOK; ++m) {
      int rm = m / 7, cm = m % 7;
      int rel = (rn - rm + 6) * 13 + (cn - cm + 6);
      float val = S[n * 68 + m] + rpb[rel * HEADS + h] + mrow[n * NTOK + m];
      S[n * 68 + m] = val;
      mx = fmaxf(mx, val);
    }
    float sum = 0.f;
    for (int m = 0; m < NTOK; ++m) {
      float e = __expf(S[n * 68 + m] - mx);
      S[n * 68 + m] = e;
      sum += e;
    }
    float inv = 1.f / sum;
    for (int m = 0; m < NTOK; ++m)
      P[n * 72 + m] = (_Float16)(S[n * 68 + m] * inv);
  }
  __syncthreads();

  // B = v tiles (K=token, N=d); padded tokens are zero in memory
  v16h bv[2][2];
#pragma unroll
  for (int nj = 0; nj < 2; ++nj)
#pragma unroll
    for (int ks = 0; ks < 2; ++ks)
#pragma unroll
      for (int r = 0; r < 8; ++r) {
        int tk = ks * 32 + koff + 2 * r;
        int d = nj * 16 + rowl;
        bv[nj][ks][2 * r]     = v[tk * HDIM + d];
        bv[nj][ks][2 * r + 1] = v[(tk + 1) * HDIM + d];
      }
  v8f o[4][2];
#pragma unroll
  for (int mi = 0; mi < 4; ++mi)
#pragma unroll
    for (int nj = 0; nj < 2; ++nj) o[mi][nj] = zero;
#pragma unroll
  for (int mi = 0; mi < 4; ++mi)
#pragma unroll
    for (int ks = 0; ks < 2; ++ks) {
      v16h ap;
#pragma unroll
      for (int r = 0; r < 8; ++r) {
        int kbx = ks * 32 + half_sel * 16 + r * 2;
        union { unsigned u; _Float16 f[2]; } cv;
        cv.u = *(const unsigned*)(P + (mi * 16 + rowl) * 72 + kbx);
        ap[2 * r] = cv.f[0];
        ap[2 * r + 1] = cv.f[1];
      }
#pragma unroll
      for (int nj = 0; nj < 2; ++nj)
        o[mi][nj] = wmma_f16(ap, bv[nj][ks], o[mi][nj]);
    }
#pragma unroll
  for (int mi = 0; mi < 4; ++mi)
#pragma unroll
    for (int nj = 0; nj < 2; ++nj)
#pragma unroll
      for (int r = 0; r < 8; ++r) {
        int tok = mi * 16 + half_sel * 8 + r;
        if (tok < NTOK) {
          int d = nj * 16 + rowl;
          out[((size_t)win * NTOK + tok) * CDIM + h * HDIM + d] = (_Float16)o[mi][nj][r];
        }
      }
}

// ---------------------------------------------------------------------------
extern "C" void kernel_launch(void* const* d_in, const int* in_sizes, int n_in,
                              void* d_out, int out_size, void* d_ws,
                              size_t ws_size, hipStream_t stream) {
  (void)in_sizes; (void)n_in; (void)out_size; (void)ws_size;
  const float* x      = (const float*)d_in[0];
  const float* amask  = (const float*)d_in[1];
  const float* qkv_w  = (const float*)d_in[2];
  const float* qkv_b  = (const float*)d_in[3];
  const float* proj_w = (const float*)d_in[4];
  const float* proj_b = (const float*)d_in[5];
  const float* rpb    = (const float*)d_in[6];
  const float* n1g    = (const float*)d_in[7];
  const float* n1b    = (const float*)d_in[8];
  const float* n2g    = (const float*)d_in[9];
  const float* n2b    = (const float*)d_in[10];
  const float* fc1_w  = (const float*)d_in[11];
  const float* fc1_b  = (const float*)d_in[12];
  const float* fc2_w  = (const float*)d_in[13];
  const float* fc2_b  = (const float*)d_in[14];

  const size_t S = (size_t)TOK * CDIM * sizeof(_Float16);          // 19.27 MB
  const size_t P = (size_t)NWIN * HEADS * PTOK * HDIM * sizeof(_Float16); // 25.17 MB
  char* ws = (char*)d_ws;
  _Float16* winb = (_Float16*)(ws);               // LN1 windows; reused as attn out
  _Float16* qbuf = (_Float16*)(ws + S);           // q (padded); reused as LN2 out
  _Float16* kbuf = (_Float16*)(ws + S + P);
  _Float16* vbuf = (_Float16*)(ws + S + 2 * P);
  float*    xr   = (float*)(ws + S + 3 * P);      // 2*S bytes (f32 residual)
  _Float16* hid  = (_Float16*)(ws + S + 3 * P + 2 * S);  // 4*S bytes

  // 0) zero padding rows of q/k/v (tokens 49..63)
  zero_pad_kernel<<<(NWIN * HEADS * 240 + 255) / 256, 256, 0, stream>>>(
      (unsigned*)qbuf, (unsigned*)kbuf, (unsigned*)vbuf);
  // 1) LN1 + shift + window partition -> f16
  ln_kernel<<<TOK, 128, 0, stream>>>(x, n1g, n1b, winb, 0);
  // 2) QKV GEMM -> q/k/v per head (padded layout)
  gemm_kernel<0><<<dim3(QKV_DIM / 128, TOK / 128), 256, 0, stream>>>(
      winb, qkv_w, qkv_b, TOK, QKV_DIM, CDIM, nullptr, nullptr, nullptr,
      qbuf, kbuf, vbuf);
  // 3) windowed attention (one wave per window-head)
  attn_kernel<<<(NWIN * HEADS) / 2, 64, 0, stream>>>(qbuf, kbuf, vbuf, rpb,
                                                     amask, winb);
  // 4) proj GEMM + window-reverse + roll + residual -> xr
  gemm_kernel<1><<<dim3(CDIM / 128, TOK / 128), 256, 0, stream>>>(
      winb, proj_w, proj_b, TOK, CDIM, CDIM, nullptr, xr, x,
      nullptr, nullptr, nullptr);
  // 5) LN2 -> f16
  ln_kernel<<<TOK, 128, 0, stream>>>(xr, n2g, n2b, qbuf, 1);
  // 6) FC1 GEMM + GELU -> f16 hid
  gemm_kernel<2><<<dim3(HID_DIM / 128, TOK / 128), 256, 0, stream>>>(
      qbuf, fc1_w, fc1_b, TOK, HID_DIM, CDIM, hid, nullptr, nullptr,
      nullptr, nullptr, nullptr);
  // 7) FC2 GEMM + residual -> d_out (f32)
  gemm_kernel<3><<<dim3(CDIM / 128, TOK / 128), 256, 0, stream>>>(
      hid, fc2_w, fc2_b, TOK, CDIM, HID_DIM, nullptr, (float*)d_out, xr,
      nullptr, nullptr, nullptr);
}
